// MultiHeadAttention_36043365548746
// MI455X (gfx1250) — matrix operand order repaired and checked by reading, not compiled, not run on hardware
//
#include <hip/hip_runtime.h>
#include <hip/hip_bf16.h>
#include <math.h>

typedef __bf16 bf16;
typedef __attribute__((ext_vector_type(16))) __bf16 v16bf;
typedef __attribute__((ext_vector_type(8)))  __bf16 v8bf;
typedef __attribute__((ext_vector_type(8)))  float  v8f;

#define D_MODEL 512
#define SEQ_L   1024
#define BATCH   8
#define HEADS   8
#define DK      64

static __device__ __forceinline__ v16bf cat8(v8bf lo, v8bf hi) {
  return __builtin_shufflevector(lo, hi, 0,1,2,3,4,5,6,7,8,9,10,11,12,13,14,15);
}

static __device__ __forceinline__ v8f wmma_bf16(v16bf a, v16bf b, v8f c) {
  return __builtin_amdgcn_wmma_f32_16x16x32_bf16(false, a, false, b, (short)0, c,
                                                 false, false);
}

// ---------------------------------------------------------------------------
// f32 -> bf16 elementwise cast (streaming: nontemporal reads)
// ---------------------------------------------------------------------------
__global__ void cvt_bf16_kernel(const float* __restrict__ x, bf16* __restrict__ y,
                                int n) {
  int i = blockIdx.x * blockDim.x + threadIdx.x;
  if (i < n) y[i] = (bf16)__builtin_nontemporal_load(&x[i]);
}

// ---------------------------------------------------------------------------
// 512x512 transpose + cast: WT[n][k] = (bf16) W[k][n]
// block (32,8), grid (16,16)
// ---------------------------------------------------------------------------
__global__ void wtrans_kernel(const float* __restrict__ W, bf16* __restrict__ WT) {
  __shared__ float t[32][33];
  const int c0 = blockIdx.x * 32;  // n tile
  const int r0 = blockIdx.y * 32;  // k tile
  const int tx = threadIdx.x, ty = threadIdx.y;
#pragma unroll
  for (int r = 0; r < 32; r += 8)
    t[ty + r][tx] = W[(size_t)(r0 + ty + r) * D_MODEL + c0 + tx];
  __syncthreads();
#pragma unroll
  for (int r = 0; r < 32; r += 8)
    WT[(size_t)(c0 + ty + r) * D_MODEL + r0 + tx] = (bf16)t[tx][ty + r];
}

// ---------------------------------------------------------------------------
// bf16 transpose per batch: vT[b][hd][l] = vw[b][l][hd]
// block (32,8), grid (512/32, 1024/32, B)
// ---------------------------------------------------------------------------
__global__ void vtrans_kernel(const bf16* __restrict__ vw, bf16* __restrict__ vT) {
  __shared__ bf16 t[32][33];
  const int b  = blockIdx.z;
  const int c0 = blockIdx.x * 32;  // hd tile
  const int l0 = blockIdx.y * 32;  // l tile
  const int tx = threadIdx.x, ty = threadIdx.y;
#pragma unroll
  for (int r = 0; r < 32; r += 8)
    t[ty + r][tx] = vw[((size_t)b * SEQ_L + l0 + ty + r) * D_MODEL + c0 + tx];
  __syncthreads();
#pragma unroll
  for (int r = 0; r < 32; r += 8)
    vT[((size_t)b * D_MODEL + c0 + ty + r) * SEQ_L + l0 + tx] = t[tx][ty + r];
}

// ---------------------------------------------------------------------------
// Projection GEMM: C(8192x512) = A(8192x512,bf16) x W(512x512) + bias
// 2x2 register blocking: one wave -> 32x32 output block (4 WMMA tiles),
// each A/B fragment reused twice -> 2 b128 loads per WMMA.
// block 256 (8 waves, 2x4 wave grid) -> 64 rows x 128 cols per block.
// grid (512/128, 8192/64). Output bf16 (outb) or f32 (outf).
// ---------------------------------------------------------------------------
__global__ void proj_gemm_kernel(const bf16* __restrict__ A,
                                 const bf16* __restrict__ WT,
                                 const float* __restrict__ bias,
                                 bf16* __restrict__ outb,
                                 float* __restrict__ outf) {
  const int tid  = threadIdx.x;
  const int wave = tid >> 5;
  const int lane = tid & 31;
  const int n    = lane & 15;
  const int hi   = lane >> 4;

  const int row0 = blockIdx.y * 64 + (wave >> 2) * 32;
  const int col0 = blockIdx.x * 128 + (wave & 3) * 32;

  const bf16* arow0 = A  + (size_t)(row0 + n) * D_MODEL;
  const bf16* arow1 = A  + (size_t)(row0 + 16 + n) * D_MODEL;
  const bf16* brow0 = WT + (size_t)(col0 + n) * D_MODEL;
  const bf16* brow1 = WT + (size_t)(col0 + 16 + n) * D_MODEL;

  v8f acc00 = {}, acc01 = {}, acc10 = {}, acc11 = {};
#pragma unroll 2
  for (int kk = 0; kk < D_MODEL; kk += 32) {
    v16bf a0 = cat8(*(const v8bf*)(arow0 + kk + hi * 8),
                    *(const v8bf*)(arow0 + kk + 16 + hi * 8));
    v16bf a1 = cat8(*(const v8bf*)(arow1 + kk + hi * 8),
                    *(const v8bf*)(arow1 + kk + 16 + hi * 8));
    v16bf b0 = cat8(*(const v8bf*)(brow0 + kk + hi * 8),
                    *(const v8bf*)(brow0 + kk + 16 + hi * 8));
    v16bf b1 = cat8(*(const v8bf*)(brow1 + kk + hi * 8),
                    *(const v8bf*)(brow1 + kk + 16 + hi * 8));
    acc00 = wmma_bf16(a0, b0, acc00);
    acc01 = wmma_bf16(a0, b1, acc01);
    acc10 = wmma_bf16(a1, b0, acc10);
    acc11 = wmma_bf16(a1, b1, acc11);
  }

  const float bv0 = bias[col0 + n];
  const float bv1 = bias[col0 + 16 + n];
  if (outf) {
#pragma unroll
    for (int i = 0; i < 8; ++i) {
      const int r0 = row0 + i + hi * 8;
      const int r1 = r0 + 16;
      __builtin_nontemporal_store(acc00[i] + bv0, &outf[(size_t)r0 * D_MODEL + col0 + n]);
      __builtin_nontemporal_store(acc01[i] + bv1, &outf[(size_t)r0 * D_MODEL + col0 + 16 + n]);
      __builtin_nontemporal_store(acc10[i] + bv0, &outf[(size_t)r1 * D_MODEL + col0 + n]);
      __builtin_nontemporal_store(acc11[i] + bv1, &outf[(size_t)r1 * D_MODEL + col0 + 16 + n]);
    }
  } else {
#pragma unroll
    for (int i = 0; i < 8; ++i) {
      const int r0 = row0 + i + hi * 8;
      const int r1 = r0 + 16;
      outb[(size_t)r0 * D_MODEL + col0 + n]      = (bf16)(acc00[i] + bv0);
      outb[(size_t)r0 * D_MODEL + col0 + 16 + n] = (bf16)(acc01[i] + bv1);
      outb[(size_t)r1 * D_MODEL + col0 + n]      = (bf16)(acc10[i] + bv0);
      outb[(size_t)r1 * D_MODEL + col0 + 16 + n] = (bf16)(acc11[i] + bv1);
    }
  }
}

// ---------------------------------------------------------------------------
// Fused attention: per (b, h, 16-query tile)
//   scores = scale * qh kh^T        (WMMA bf16 -> f32, into f32 LDS strip)
//   attn   = softmax(scores)        (f32 in LDS; written once, nontemporal,
//                                    to d_out; bf16 copy kept in LDS)
//   ctx    = attn vh                (WMMA; A operand = bf16 LDS b128 loads)
// block 256 (8 waves). Both LDS strips padded for conflict-free b128 access.
// ---------------------------------------------------------------------------
__global__ void attn_fused_kernel(const bf16* __restrict__ qw,
                                  const bf16* __restrict__ kw,
                                  const bf16* __restrict__ vT,
                                  float* __restrict__ attn_out,
                                  bf16* __restrict__ ctx_out) {
  __shared__ float sc[16][SEQ_L + 4];   // raw scores / exp (f32)
  __shared__ bf16  scb[16][SEQ_L + 8];  // normalized probabilities (bf16)
  __shared__ float red[256];
  __shared__ float rowstat[16];
  __shared__ float ctxbuf[4][8][32];

  const int b  = blockIdx.z;
  const int h  = blockIdx.y;
  const int q0 = blockIdx.x * 16;

  const int tid  = threadIdx.x;
  const int wave = tid >> 5;
  const int lane = tid & 31;
  const int n    = lane & 15;
  const int hi   = lane >> 4;

  // ---- Phase 1: scores strip (16 queries x 1024 keys), wave w -> keys [w*128,+128)
  const bf16* qrow = qw + ((size_t)(b * SEQ_L + q0 + n)) * D_MODEL + h * DK;
  v16bf aq0 = cat8(*(const v8bf*)(qrow + hi * 8),
                   *(const v8bf*)(qrow + 16 + hi * 8));
  v16bf aq1 = cat8(*(const v8bf*)(qrow + 32 + hi * 8),
                   *(const v8bf*)(qrow + 48 + hi * 8));
#pragma unroll
  for (int t = 0; t < 8; ++t) {
    const int key0 = (wave * 8 + t) * 16;
    const bf16* krow = kw + ((size_t)(b * SEQ_L + key0 + n)) * D_MODEL + h * DK;
    v16bf bb0 = cat8(*(const v8bf*)(krow + hi * 8),
                     *(const v8bf*)(krow + 16 + hi * 8));
    v16bf bb1 = cat8(*(const v8bf*)(krow + 32 + hi * 8),
                     *(const v8bf*)(krow + 48 + hi * 8));
    v8f acc = {};
    acc = wmma_bf16(aq0, bb0, acc);
    acc = wmma_bf16(aq1, bb1, acc);
#pragma unroll
    for (int i = 0; i < 8; ++i)
      sc[i + hi * 8][key0 + n] = acc[i] * 0.125f;  // 1/sqrt(64)
  }
  __syncthreads();

  // ---- Phase 2: row softmax (16 threads per row)
  const int row = tid >> 4;
  const int c0  = tid & 15;
  float m = -3.0e38f;
#pragma unroll 8
  for (int j = 0; j < 64; ++j) m = fmaxf(m, sc[row][c0 + j * 16]);
  red[tid] = m;
  __syncthreads();
  if (c0 == 0) {
    float mm = red[row * 16];
    for (int t = 1; t < 16; ++t) mm = fmaxf(mm, red[row * 16 + t]);
    rowstat[row] = mm;
  }
  __syncthreads();
  const float rmax = rowstat[row];
  float s = 0.0f;
#pragma unroll 8
  for (int j = 0; j < 64; ++j) {
    const float e = __expf(sc[row][c0 + j * 16] - rmax);
    sc[row][c0 + j * 16] = e;
    s += e;
  }
  red[tid] = s;
  __syncthreads();
  if (c0 == 0) {
    float ss = 0.0f;
    for (int t = 0; t < 16; ++t) ss += red[row * 16 + t];
    rowstat[row] = ss;
  }
  __syncthreads();
  const float rinv = 1.0f / rowstat[row];
  float* arow = attn_out + ((size_t)((b * HEADS + h) * SEQ_L + q0 + row)) * SEQ_L;
#pragma unroll 8
  for (int j = 0; j < 64; ++j) {
    const int c = c0 + j * 16;
    const float p = sc[row][c] * rinv;
    scb[row][c] = (bf16)p;                      // bf16 copy for ctx WMMA
    __builtin_nontemporal_store(p, &arow[c]);   // single 268 MB attn write
  }
  __syncthreads();

  // ---- Phase 3: ctx = attn @ vh. wave -> d-tile (wave&3), K half (wave>>2)
  const int nt     = wave & 3;
  const int khalf  = (wave >> 2) * (SEQ_L / 2);
  const bf16* vrow = vT + ((size_t)(b * D_MODEL + h * DK + nt * 16 + n)) * SEQ_L;
  v8f acc = {};
#pragma unroll 2
  for (int kk = khalf; kk < khalf + SEQ_L / 2; kk += 32) {
    v16bf a = cat8(*(const v8bf*)(&scb[n][kk + hi * 8]),
                   *(const v8bf*)(&scb[n][kk + 16 + hi * 8]));
    v16bf bb = cat8(*(const v8bf*)(vrow + kk + hi * 8),
                    *(const v8bf*)(vrow + kk + 16 + hi * 8));
    acc = wmma_bf16(a, bb, acc);
  }
  if (wave >= 4) {
#pragma unroll
    for (int i = 0; i < 8; ++i) ctxbuf[wave - 4][i][lane] = acc[i];
  }
  __syncthreads();
  if (wave < 4) {
#pragma unroll
    for (int i = 0; i < 8; ++i) {
      const float v     = acc[i] + ctxbuf[wave][i][lane];
      const int  qrow_i = q0 + i + hi * 8;
      ctx_out[((size_t)(b * SEQ_L + qrow_i)) * D_MODEL + h * DK + nt * 16 + n] =
          (bf16)v;
    }
  }
}

// ---------------------------------------------------------------------------
extern "C" void kernel_launch(void* const* d_in, const int* in_sizes, int n_in,
                              void* d_out, int out_size, void* d_ws, size_t ws_size,
                              hipStream_t stream) {
  const float* q  = (const float*)d_in[0];
  const float* k  = (const float*)d_in[1];
  // d_in[2] (v) intentionally unused: reference applies dense_v to q.
  const float* Wq = (const float*)d_in[3];
  const float* bq = (const float*)d_in[4];
  const float* Wk = (const float*)d_in[5];
  const float* bk = (const float*)d_in[6];
  const float* Wv = (const float*)d_in[7];
  const float* bv = (const float*)d_in[8];
  const float* Wo = (const float*)d_in[9];
  const float* bo = (const float*)d_in[10];

  float* out  = (float*)d_out;
  float* attn = out + (size_t)BATCH * SEQ_L * D_MODEL;

  const size_t NA = (size_t)BATCH * SEQ_L * D_MODEL;  // 4,194,304 activations
  const size_t NW = (size_t)D_MODEL * D_MODEL;        // 262,144 weights

  bf16* qbf = (bf16*)d_ws;
  bf16* kbf = qbf + NA;
  bf16* WqT = kbf + NA;
  bf16* WkT = WqT + NW;
  bf16* WvT = WkT + NW;
  bf16* WoT = WvT + NW;
  bf16* qw  = WoT + NW;
  bf16* kw  = qw + NA;
  bf16* vw  = kw + NA;
  bf16* vT  = vw + NA;
  bf16* ctx = vT + NA;
  (void)in_sizes; (void)n_in; (void)out_size; (void)ws_size;

  // 1) casts
  cvt_bf16_kernel<<<(int)(NA / 256), 256, 0, stream>>>(q, qbf, (int)NA);
  cvt_bf16_kernel<<<(int)(NA / 256), 256, 0, stream>>>(k, kbf, (int)NA);

  // 2) weight transposes (f32 -> bf16 W^T)
  dim3 tb(32, 8);
  wtrans_kernel<<<dim3(16, 16), tb, 0, stream>>>(Wq, WqT);
  wtrans_kernel<<<dim3(16, 16), tb, 0, stream>>>(Wk, WkT);
  wtrans_kernel<<<dim3(16, 16), tb, 0, stream>>>(Wv, WvT);
  wtrans_kernel<<<dim3(16, 16), tb, 0, stream>>>(Wo, WoT);

  // 3) projections (note vw uses qbf: faithful to reference)
  dim3 pg(D_MODEL / 128, (BATCH * SEQ_L) / 64);
  proj_gemm_kernel<<<pg, 256, 0, stream>>>(qbf, WqT, bq, qw, nullptr);
  proj_gemm_kernel<<<pg, 256, 0, stream>>>(kbf, WkT, bk, kw, nullptr);
  proj_gemm_kernel<<<pg, 256, 0, stream>>>(qbf, WvT, bv, vw, nullptr);

  // 4) value transpose for K-contiguous WMMA B loads
  vtrans_kernel<<<dim3(D_MODEL / 32, SEQ_L / 32, BATCH), tb, 0, stream>>>(vw, vT);

  // 5) fused scores -> softmax -> attn write -> ctx
  attn_fused_kernel<<<dim3(SEQ_L / 16, HEADS, BATCH), 256, 0, stream>>>(
      qw, kw, vT, attn, ctx);

  // 6) output projection (f32 out + bias)
  proj_gemm_kernel<<<pg, 256, 0, stream>>>(ctx, WoT, bo, nullptr, out);
}
